// GCN_node_73375221285623
// MI455X (gfx1250) — matrix operand order repaired and checked by reading, not compiled, not run on hardware
//
#include <hip/hip_runtime.h>

// ---------------------------------------------------------------------------
// CDNA5 / gfx1250 implicit-GNN (INECN) solver.
// All 64-deep contractions run on V_WMMA_F32_16X16X32_BF16 (2 wmma per 16x16
// C-tile). State X and edge-aggregate are kept transposed [N][64] so gathers
// are line-contiguous and WMMA B-fragments are single 32B aligned loads.
// ---------------------------------------------------------------------------

typedef __attribute__((ext_vector_type(16))) __bf16 bf16x16;  // 32 B, 8 VGPRs
typedef __attribute__((ext_vector_type(8)))  float  f32x8;    // C/D accumulator

static_assert(sizeof(bf16x16) == 32, "bf16x16 size");

union AFrag { bf16x16 v; uint4 q[2]; };
union PackB { uint4 q; __bf16 h[8]; };

__device__ __forceinline__ f32x8 wmma_step(bf16x16 a, bf16x16 b, f32x8 c) {
    // D = A(16x32 bf16) * B(32x16 bf16) + C(16x16 f32)
    return __builtin_amdgcn_wmma_f32_16x16x32_bf16(
        /*neg_a=*/false, a, /*neg_b=*/false, b,
        /*c_mod=*/(short)0, c, /*reuse_a=*/false, /*reuse_b=*/false);
}

// A-matrix fragment (16-bit A 16x32): lane m=l&15, half=l>>4;
// element i <-> K = kbase + 16*half + i (contiguous, matches B fragment)
__device__ __forceinline__ bf16x16 load_a_frag(const __bf16* __restrict__ W,
                                               int ldk, int row0, int kbase, int lane) {
    const int m  = lane & 15;
    const int hh = lane >> 4;
    const __bf16* p = W + (size_t)(row0 + m) * ldk + kbase + 16 * hh;
    AFrag f;
    f.q[0] = *(const uint4*)(p);        // K-run 0..7
    f.q[1] = *(const uint4*)(p + 8);    // K-run 8..15
    return f.v;
}

// B-matrix fragment (dense 16-bit B 32x16, contiguous k order, same as load_a_frag):
// lane n=l&15, half=l>>4; element i <-> B[kbase + 16*half + i][n].
// 'col' points at a K-contiguous column (row of the transposed [cols][K] array).
__device__ __forceinline__ bf16x16 load_b_frag(const __bf16* __restrict__ col,
                                               int kbase, int lane) {
    const int hh = lane >> 4;
    return *(const bf16x16*)(col + kbase + 16 * hh);   // 32 B aligned
}

__device__ __forceinline__ bf16x16 load_b_frag_f32(const float* __restrict__ col,
                                                   int kbase, int lane) {
    const int hh = lane >> 4;
    const float* p = col + kbase + 16 * hh;
    bf16x16 r;
#pragma unroll
    for (int i = 0; i < 16; ++i) r[i] = (__bf16)p[i];
    return r;
}

// ---------------------------------------------------------------------------
// Small prep kernels
// ---------------------------------------------------------------------------

__global__ void zero_kernel(unsigned* __restrict__ p, long n) {
    long i = (long)blockIdx.x * blockDim.x + threadIdx.x;
    if (i < n) p[i] = 0u;
}

__global__ void convert_bf16_kernel(const float* __restrict__ in,
                                    __bf16* __restrict__ out, int n) {
    int i = blockIdx.x * blockDim.x + threadIdx.x;
    if (i < n) out[i] = (__bf16)in[i];
}

// [F][C] f32 -> [C][F] bf16 (one-shot, input-bound; simplicity over peak BW)
__global__ void transpose_bf16_kernel(const float* __restrict__ in,
                                      __bf16* __restrict__ out, int F, long C) {
    long idx = (long)blockIdx.x * blockDim.x + threadIdx.x;
    long total = (long)F * C;
    if (idx >= total) return;
    int  f = (int)(idx % F);
    long c = idx / F;
    out[idx] = (__bf16)in[(size_t)f * C + c];
}

// Wn_hat = kappa * Wn / max(frob(Wn), 1), converted to bf16. Single block.
__global__ void wn_norm_scale_kernel(const float* __restrict__ Wn,
                                     __bf16* __restrict__ out) {
    __shared__ float red[256];
    __shared__ float scale;
    float s = 0.f;
    for (int i = threadIdx.x; i < 4096; i += 256) { float v = Wn[i]; s += v * v; }
    red[threadIdx.x] = s;
    __syncthreads();
    for (int off = 128; off > 0; off >>= 1) {
        if ((int)threadIdx.x < off) red[threadIdx.x] += red[threadIdx.x + off];
        __syncthreads();
    }
    if (threadIdx.x == 0) scale = 0.9f / fmaxf(sqrtf(red[0]), 1.0f);
    __syncthreads();
    for (int i = threadIdx.x; i < 4096; i += 256) out[i] = (__bf16)(Wn[i] * scale);
}

// ---------------------------------------------------------------------------
// C[64][cols] = A[64][K] @ Bt (Bt stored [cols][K] bf16), K in {32, 64}.
// Used for bias_n = Omega_n @ node_data and bias_e = Omega_e @ Ra_data.
// ---------------------------------------------------------------------------
__global__ __launch_bounds__(128) void gemm_bias_kernel(
    const __bf16* __restrict__ A, const __bf16* __restrict__ Bt,
    float* __restrict__ C, int K, long cols, int numTiles, int tilesPerBlock)
{
    const int lane = threadIdx.x & 31;
    const int w    = threadIdx.x >> 5;   // 16-row strip
    const int hh   = lane >> 4;
    const int nl   = lane & 15;

    const bf16x16 a0 = load_a_frag(A, K, 16 * w, 0, lane);
    bf16x16 a1 = a0;
    if (K == 64) a1 = load_a_frag(A, K, 16 * w, 32, lane);

    for (int t = 0; t < tilesPerBlock; ++t) {
        const int tile = blockIdx.x * tilesPerBlock + t;
        if (tile >= numTiles) return;                 // block-uniform
        const long n = (long)tile * 16 + nl;

        f32x8 acc = {};
        bf16x16 b0 = load_b_frag(Bt + (size_t)n * K, 0, lane);
        acc = wmma_step(a0, b0, acc);
        if (K == 64) {
            bf16x16 b1 = load_b_frag(Bt + (size_t)n * K, 32, lane);
            acc = wmma_step(a1, b1, acc);
        }
#pragma unroll
        for (int v = 0; v < 8; ++v)
            C[(size_t)(16 * w + 8 * hh + v) * cols + n] = acc[v];
    }
}

// ---------------------------------------------------------------------------
// Edge pass: He = relu(W_es @ X[:,S] + W_er @ X[:,R] + bias_e).
// mode 0: atomically scatter He columns into aggT[N][64].
// mode 1: write He and pre-relu logits ([64][E] row-major) to output.
// One wave per (16-edge tile, 16-row strip); A-fragments live in VGPRs.
// ---------------------------------------------------------------------------
__global__ __launch_bounds__(128) void edge_kernel(
    const __bf16* __restrict__ Xb,      // [N][64] bf16 state
    const __bf16* __restrict__ Wes, const __bf16* __restrict__ Wer,
    const float*  __restrict__ biasE,   // [64][E] f32
    const int* __restrict__ S, const int* __restrict__ R, const int* __restrict__ H,
    float* __restrict__ aggT,           // [N][64] f32 (mode 0)
    float* __restrict__ heOut, float* __restrict__ lgOut,   // [64][E] (mode 1)
    int mode, long E, int numTiles, int tilesPerBlock)
{
    const int lane = threadIdx.x & 31;
    const int w    = threadIdx.x >> 5;
    const int hh   = lane >> 4;
    const int nl   = lane & 15;

    const bf16x16 aes0 = load_a_frag(Wes, 64, 16 * w, 0,  lane);
    const bf16x16 aes1 = load_a_frag(Wes, 64, 16 * w, 32, lane);
    const bf16x16 aer0 = load_a_frag(Wer, 64, 16 * w, 0,  lane);
    const bf16x16 aer1 = load_a_frag(Wer, 64, 16 * w, 32, lane);

    for (int t = 0; t < tilesPerBlock; ++t) {
        const int tile = blockIdx.x * tilesPerBlock + t;
        if (tile >= numTiles) return;                 // block-uniform
        const long e = (long)tile * 16 + nl;
        const int  s = S[e];
        const int  r = R[e];

        // gathered node columns are contiguous 128 B lines
        const bf16x16 bs0 = load_b_frag(Xb + (size_t)s * 64, 0,  lane);
        const bf16x16 bs1 = load_b_frag(Xb + (size_t)s * 64, 32, lane);
        const bf16x16 br0 = load_b_frag(Xb + (size_t)r * 64, 0,  lane);
        const bf16x16 br1 = load_b_frag(Xb + (size_t)r * 64, 32, lane);

        f32x8 acc;
#pragma unroll
        for (int v = 0; v < 8; ++v)
            acc[v] = biasE[(size_t)(16 * w + 8 * hh + v) * E + e];

        acc = wmma_step(aes0, bs0, acc);
        acc = wmma_step(aes1, bs1, acc);
        acc = wmma_step(aer0, br0, acc);
        acc = wmma_step(aer1, br1, acc);

        if (mode == 0) {
            const int h = H[e];
            float* dst = aggT + (size_t)h * 64 + 16 * w + 8 * hh;
#pragma unroll
            for (int v = 0; v < 8; ++v)
                atomicAdd(dst + v, fmaxf(acc[v], 0.0f));
        } else {
#pragma unroll
            for (int v = 0; v < 8; ++v) {
                const size_t idx = (size_t)(16 * w + 8 * hh + v) * E + e;
                lgOut[idx] = acc[v];
                heOut[idx] = fmaxf(acc[v], 0.0f);
            }
        }
    }
}

// ---------------------------------------------------------------------------
// Node pass: X' = relu(Wn_hat @ X + Bm @ agg + bias_n); writes bf16 [N][64].
// ---------------------------------------------------------------------------
__global__ __launch_bounds__(128) void node_kernel(
    const __bf16* __restrict__ Xin, const float* __restrict__ aggT,
    const float* __restrict__ biasN, const __bf16* __restrict__ WnHat,
    const __bf16* __restrict__ Bm, __bf16* __restrict__ Xout,
    long N, int numTiles, int tilesPerBlock)
{
    const int lane = threadIdx.x & 31;
    const int w    = threadIdx.x >> 5;
    const int hh   = lane >> 4;
    const int nl   = lane & 15;

    const bf16x16 awn0 = load_a_frag(WnHat, 64, 16 * w, 0,  lane);
    const bf16x16 awn1 = load_a_frag(WnHat, 64, 16 * w, 32, lane);
    const bf16x16 abm0 = load_a_frag(Bm,    64, 16 * w, 0,  lane);
    const bf16x16 abm1 = load_a_frag(Bm,    64, 16 * w, 32, lane);

    for (int t = 0; t < tilesPerBlock; ++t) {
        const int tile = blockIdx.x * tilesPerBlock + t;
        if (tile >= numTiles) return;
        const long n = (long)tile * 16 + nl;

        const bf16x16 bx0 = load_b_frag(Xin + (size_t)n * 64, 0,  lane);
        const bf16x16 bx1 = load_b_frag(Xin + (size_t)n * 64, 32, lane);
        const bf16x16 ba0 = load_b_frag_f32(aggT + (size_t)n * 64, 0,  lane);
        const bf16x16 ba1 = load_b_frag_f32(aggT + (size_t)n * 64, 32, lane);

        f32x8 acc;
#pragma unroll
        for (int v = 0; v < 8; ++v)
            acc[v] = biasN[(size_t)(16 * w + 8 * hh + v) * N + n];

        acc = wmma_step(awn0, bx0, acc);
        acc = wmma_step(awn1, bx1, acc);
        acc = wmma_step(abm0, ba0, acc);
        acc = wmma_step(abm1, ba1, acc);

        // D rows are v + 8*half -> 8 consecutive bf16 = one 16 B store
        PackB pk;
#pragma unroll
        for (int v = 0; v < 8; ++v) pk.h[v] = (__bf16)fmaxf(acc[v], 0.0f);
        *(uint4*)(Xout + (size_t)n * 64 + 16 * w + 8 * hh) = pk.q;
    }
}

// ---------------------------------------------------------------------------
// Dense head: out[n][:] = V1 @ relu(V0 @ X[:,n] + b0) + b1 (row-major [N][64]).
// Intermediate strip staged through 2 KB LDS between the two WMMA GEMMs.
// ---------------------------------------------------------------------------
__global__ __launch_bounds__(128) void head_kernel(
    const __bf16* __restrict__ Xb, const __bf16* __restrict__ V0,
    const __bf16* __restrict__ V1, const float* __restrict__ b0,
    const float* __restrict__ b1, float* __restrict__ outX)
{
    __shared__ __attribute__((aligned(32))) __bf16 ldsH[16 * 64];

    const int lane = threadIdx.x & 31;
    const int w    = threadIdx.x >> 5;
    const int hh   = lane >> 4;
    const int nl   = lane & 15;
    const long n   = (long)blockIdx.x * 16 + nl;

    // stage 1: h = relu(V0 @ x + b0)
    const bf16x16 a0 = load_a_frag(V0, 64, 16 * w, 0,  lane);
    const bf16x16 a1 = load_a_frag(V0, 64, 16 * w, 32, lane);
    const bf16x16 bx0 = load_b_frag(Xb + (size_t)n * 64, 0,  lane);
    const bf16x16 bx1 = load_b_frag(Xb + (size_t)n * 64, 32, lane);

    f32x8 acc;
#pragma unroll
    for (int v = 0; v < 8; ++v) acc[v] = b0[16 * w + 8 * hh + v];
    acc = wmma_step(a0, bx0, acc);
    acc = wmma_step(a1, bx1, acc);

    PackB pk;
#pragma unroll
    for (int v = 0; v < 8; ++v) pk.h[v] = (__bf16)fmaxf(acc[v], 0.0f);
    *(uint4*)(&ldsH[nl * 64 + 16 * w + 8 * hh]) = pk.q;
    __syncthreads();

    // stage 2: y = V1 @ h + b1
    const bf16x16 c0 = load_a_frag(V1, 64, 16 * w, 0,  lane);
    const bf16x16 c1 = load_a_frag(V1, 64, 16 * w, 32, lane);
    const bf16x16 bh0 = load_b_frag(&ldsH[nl * 64], 0,  lane);
    const bf16x16 bh1 = load_b_frag(&ldsH[nl * 64], 32, lane);

    f32x8 acc2;
#pragma unroll
    for (int v = 0; v < 8; ++v) acc2[v] = b1[16 * w + 8 * hh + v];
    acc2 = wmma_step(c0, bh0, acc2);
    acc2 = wmma_step(c1, bh1, acc2);

    float* dst = outX + (size_t)n * 64 + 16 * w + 8 * hh;   // 8 consecutive f32
    *(float4*)dst       = make_float4(acc2[0], acc2[1], acc2[2], acc2[3]);
    *((float4*)dst + 1) = make_float4(acc2[4], acc2[5], acc2[6], acc2[7]);
}

// ---------------------------------------------------------------------------
// Host orchestration
// ---------------------------------------------------------------------------
extern "C" void kernel_launch(void* const* d_in, const int* in_sizes, int n_in,
                              void* d_out, int out_size, void* d_ws, size_t ws_size,
                              hipStream_t stream) {
    (void)in_sizes; (void)n_in; (void)out_size; (void)ws_size;

    const long N = 20000, E = 160000;
    const int ITERS = 8;

    const int*   R       = (const int*)d_in[0];
    const int*   S       = (const int*)d_in[1];
    const int*   H       = (const int*)d_in[2];
    const float* nodeD   = (const float*)d_in[3];   // [64][N]
    const float* RaD     = (const float*)d_in[4];   // [32][E]
    const float* W_es    = (const float*)d_in[5];
    const float* W_er    = (const float*)d_in[6];
    const float* Omega_e = (const float*)d_in[7];   // [64][32]
    const float* Wn      = (const float*)d_in[8];
    const float* Bm      = (const float*)d_in[9];
    const float* Omega_n = (const float*)d_in[10];
    const float* V0w     = (const float*)d_in[11];
    const float* V0b     = (const float*)d_in[12];
    const float* V1w     = (const float*)d_in[13];
    const float* V1b     = (const float*)d_in[14];

    // workspace carve-up (256 B aligned), total ~69 MB
    char* ws = (char*)d_ws;
    size_t off = 0;
    auto carve = [&](size_t bytes) -> void* {
        void* p = ws + off;
        off += (bytes + 255) & ~(size_t)255;
        return p;
    };
    __bf16* wesB = (__bf16*)carve(4096 * 2);
    __bf16* werB = (__bf16*)carve(4096 * 2);
    __bf16* wnB  = (__bf16*)carve(4096 * 2);
    __bf16* bmB  = (__bf16*)carve(4096 * 2);
    __bf16* omnB = (__bf16*)carve(4096 * 2);
    __bf16* omeB = (__bf16*)carve(2048 * 2);
    __bf16* v0B  = (__bf16*)carve(4096 * 2);
    __bf16* v1B  = (__bf16*)carve(4096 * 2);
    __bf16* xnT  = (__bf16*)carve((size_t)N * 64 * 2);   // node_data^T bf16
    __bf16* raT  = (__bf16*)carve((size_t)E * 32 * 2);   // Ra_data^T bf16
    float*  bN   = (float*)carve((size_t)64 * N * 4);    // bias_n [64][N]
    float*  bE   = (float*)carve((size_t)64 * E * 4);    // bias_e [64][E]
    __bf16* xb0  = (__bf16*)carve((size_t)N * 64 * 2);   // state ping
    __bf16* xb1  = (__bf16*)carve((size_t)N * 64 * 2);   // state pong
    float*  aggT = (float*)carve((size_t)N * 64 * 4);    // edge aggregate [N][64]

    auto blks = [](long n, int bs) { return (unsigned)((n + bs - 1) / bs); };

    // --- prep ---------------------------------------------------------------
    wn_norm_scale_kernel<<<1, 256, 0, stream>>>(Wn, wnB);
    convert_bf16_kernel<<<blks(4096, 256), 256, 0, stream>>>(W_es, wesB, 4096);
    convert_bf16_kernel<<<blks(4096, 256), 256, 0, stream>>>(W_er, werB, 4096);
    convert_bf16_kernel<<<blks(4096, 256), 256, 0, stream>>>(Bm, bmB, 4096);
    convert_bf16_kernel<<<blks(4096, 256), 256, 0, stream>>>(Omega_n, omnB, 4096);
    convert_bf16_kernel<<<blks(2048, 256), 256, 0, stream>>>(Omega_e, omeB, 2048);
    convert_bf16_kernel<<<blks(4096, 256), 256, 0, stream>>>(V0w, v0B, 4096);
    convert_bf16_kernel<<<blks(4096, 256), 256, 0, stream>>>(V1w, v1B, 4096);

    transpose_bf16_kernel<<<blks(64 * N, 256), 256, 0, stream>>>(nodeD, xnT, 64, N);
    transpose_bf16_kernel<<<blks(32 * E, 256), 256, 0, stream>>>(RaD, raT, 32, E);

    zero_kernel<<<blks(N * 32, 256), 256, 0, stream>>>((unsigned*)xb0, N * 32);  // X0 = 0

    // bias_n = Omega_n @ node_data ; bias_e = Omega_e @ Ra_data
    const int nTilesN = (int)(N / 16), nTilesE = (int)(E / 16), TPB = 4;
    gemm_bias_kernel<<<blks(nTilesN, TPB), 128, 0, stream>>>(omnB, xnT, bN, 64, N, nTilesN, TPB);
    gemm_bias_kernel<<<blks(nTilesE, TPB), 128, 0, stream>>>(omeB, raT, bE, 32, E, nTilesE, TPB);

    // --- 8 Picard iterations ------------------------------------------------
    for (int it = 0; it < ITERS; ++it) {
        const __bf16* xin = (it & 1) ? xb1 : xb0;
        __bf16* xout      = (it & 1) ? xb0 : xb1;
        zero_kernel<<<blks(N * 64, 256), 256, 0, stream>>>((unsigned*)aggT, N * 64);
        edge_kernel<<<blks(nTilesE, TPB), 128, 0, stream>>>(
            xin, wesB, werB, bE, S, R, H, aggT, nullptr, nullptr,
            /*mode=*/0, E, nTilesE, TPB);
        node_kernel<<<blks(nTilesN, TPB), 128, 0, stream>>>(
            xin, aggT, bN, wnB, bmB, xout, N, nTilesN, TPB);
    }
    // after 8 iterations the fixed point lives in xb0

    // --- outputs: (x [N][64], He [64][E], logits [64][E]) -------------------
    float* out    = (float*)d_out;
    float* outX   = out;
    float* outHe  = out + (size_t)N * 64;
    float* outLg  = outHe + (size_t)64 * E;

    edge_kernel<<<blks(nTilesE, TPB), 128, 0, stream>>>(
        xb0, wesB, werB, bE, S, R, H, nullptr, outHe, outLg,
        /*mode=*/1, E, nTilesE, TPB);
    head_kernel<<<(unsigned)(N / 16), 128, 0, stream>>>(xb0, v0B, v1B, V0b, V1b, outX);
}
